// TextureExtractor1_32504312496378
// MI455X (gfx1250) — matrix operand order repaired and checked by reading, not compiled, not run on hardware
//
#include <hip/hip_runtime.h>
#include <hip/hip_bf16.h>
#include <stdint.h>

typedef __attribute__((ext_vector_type(16))) _Float16 v16h;
typedef __attribute__((ext_vector_type(8)))  float    v8f;
// 4-byte-aligned float4 (neighbor bases are only dword aligned due to dx=+/-5)
typedef float floatx4 __attribute__((ext_vector_type(4), aligned(4)));

#define IMG_W  1024
#define IMG_H  1024
#define NPIX   (IMG_W * IMG_H)
#define NIMG   8
#define NOFF   4
#define LEVELS 16
#define MMB    32   // min/max blocks per image
#define BPI    64   // GLCM blocks per image (x 8 waves = 512 waves/image)

// ---- monotone float <-> ordered-uint key (handles negatives) ----
__device__ __forceinline__ uint32_t f2key(float f) {
  uint32_t u = __float_as_uint(f);
  return (u & 0x80000000u) ? ~u : (u | 0x80000000u);
}
__device__ __forceinline__ float key2f(uint32_t k) {
  uint32_t u = (k & 0x80000000u) ? (k ^ 0x80000000u) : ~k;
  return __uint_as_float(u);
}

// ---------------- kernel 0: re-init workspace every call ----------------
__global__ void init_ws_kernel(uint32_t* keys, float* nacc, float* pcacc) {
  int t = threadIdx.x;
  if (t < NIMG) { keys[2 * t] = 0xFFFFFFFFu; keys[2 * t + 1] = 0u; }
  for (int i = t; i < NIMG * NOFF * 256; i += 256) nacc[i] = 0.0f;
  for (int i = t; i < NIMG * NOFF * 16;  i += 256) pcacc[i] = 0.0f;
}

// ---------------- kernel 1: per-image min / max ----------------
__global__ __launch_bounds__(256) void minmax_kernel(const float* __restrict__ x,
                                                     uint32_t* __restrict__ keys) {
  const int img = blockIdx.x / MMB;
  const int blk = blockIdx.x % MMB;
  const floatx4* p = (const floatx4*)(x + (size_t)img * NPIX);
  float mn = 3.402823466e38f, mx = -3.402823466e38f;
  for (int i = blk * 256 + threadIdx.x; i < NPIX / 4; i += MMB * 256) {
    floatx4 v = p[i];
    mn = fminf(mn, fminf(fminf(v.x, v.y), fminf(v.z, v.w)));
    mx = fmaxf(mx, fmaxf(fmaxf(v.x, v.y), fmaxf(v.z, v.w)));
  }
  __shared__ float smn[256], smx[256];
  smn[threadIdx.x] = mn; smx[threadIdx.x] = mx;
  __syncthreads();
  for (int s = 128; s > 0; s >>= 1) {
    if (threadIdx.x < s) {
      smn[threadIdx.x] = fminf(smn[threadIdx.x], smn[threadIdx.x + s]);
      smx[threadIdx.x] = fmaxf(smx[threadIdx.x], smx[threadIdx.x + s]);
    }
    __syncthreads();
  }
  if (threadIdx.x == 0) {
    atomicMin(&keys[2 * img],     f2key(smn[0]));
    atomicMax(&keys[2 * img + 1], f2key(smx[0]));
  }
}

// ---------------- kernel 2: WMMA GLCM accumulation ----------------
// One wave owns 4 GLCM f32 accumulators (one per offset). A 32-pixel row chunk
// feeds one V_WMMA_F32_16X16X32_F16 per offset:
//   A[m][k] = (q(center_k) == m+1)                    (shared by all offsets)
//   B[k][b] = (q(neighbor_k) == b+1) * valid_k        (per offset)
// Pixel->K bijection chosen so each half-wave only needs half-wave-uniform
// contiguous loads (per ISA A-layout: lanes 0-15 own K{0..7,16..23}).
__global__ __launch_bounds__(256) void glcm_wmma_kernel(const float* __restrict__ x,
                                                        const uint32_t* __restrict__ keys,
                                                        float* __restrict__ nacc,
                                                        float* __restrict__ pcacc) {
  constexpr int DY[4] = {0, -5, -5, -5};
  constexpr int DX[4] = {5,  5,  0, -5};

  const int img  = blockIdx.x / BPI;
  const int blk  = blockIdx.x % BPI;
  const int lane = threadIdx.x & 31;
  const int wav  = threadIdx.x >> 5;
  const int gw   = blk * 8 + wav;   // 0..BPI*8-1 within image
  const int TW   = BPI * 8;         // waves per image

  const float* im = x + (size_t)img * NPIX;
  const float mn = key2f(keys[2 * img]);
  const float mx = key2f(keys[2 * img + 1]);
  const float scale = (float)(LEVELS - 1) / (mx - mn);
  const float beta  = 1.0f - mn * scale;      // q = fma(v, scale, beta)

  const int   g   = lane >> 4;      // half-wave group
  const int   m   = lane & 15;      // level / column index owned by this lane
  const float tgt = (float)(m + 1);

  v8f zero = {};
  v8f acc[4];
#pragma unroll
  for (int o = 0; o < 4; ++o) acc[o] = zero;
  float pc[4] = {0.f, 0.f, 0.f, 0.f};

  const int CHUNKS = NPIX / 32;     // 32 pixels per chunk, chunks stay in-row
  for (int c = gw; c < CHUNKS; c += TW) {
    const int y  = c >> 5;
    const int x0 = (c & 31) << 5;
    const int rowbase = y * IMG_W + x0;

    // ---- load center pixels for this half-wave's A rows ----
    float av[16];
    {
      const float* pa = im + rowbase + 8 * g;    // 32B aligned
      floatx4 a0 = *(const floatx4*)(pa + 0);
      floatx4 a1 = *(const floatx4*)(pa + 4);
      floatx4 a2 = *(const floatx4*)(pa + 16);
      floatx4 a3 = *(const floatx4*)(pa + 20);
      av[0]=a0.x; av[1]=a0.y; av[2]=a0.z;  av[3]=a0.w;
      av[4]=a1.x; av[5]=a1.y; av[6]=a1.z;  av[7]=a1.w;
      av[8]=a2.x; av[9]=a2.y; av[10]=a2.z; av[11]=a2.w;
      av[12]=a3.x; av[13]=a3.y; av[14]=a3.z; av[15]=a3.w;
    }

    int interior = (y >= 5) & (x0 != 0) & (x0 != IMG_W - 32);
    interior = __builtin_amdgcn_readfirstlane(interior);  // scalar branch, EXEC stays full

    if (interior) {
      // ---- A fragment (no pad bookkeeping needed) ----
      v16h A;
#pragma unroll
      for (int s = 0; s < 16; ++s) {
        float q = fmaf(av[s], scale, beta);
        A[s] = (q == tgt) ? (_Float16)1.0f : (_Float16)0.0f;
      }
#pragma unroll
      for (int o = 0; o < 4; ++o) {
        const int offlin = DY[o] * IMG_W + DX[o];
        const float* nb = im + rowbase + offlin + 8 * g;
        floatx4 w0 = *(const floatx4*)(nb + 0);
        floatx4 w1 = *(const floatx4*)(nb + 4);
        floatx4 w2 = *(const floatx4*)(nb + 16);
        floatx4 w3 = *(const floatx4*)(nb + 20);
        float bv[16];
        bv[0]=w0.x; bv[1]=w0.y; bv[2]=w0.z;  bv[3]=w0.w;
        bv[4]=w1.x; bv[5]=w1.y; bv[6]=w1.z;  bv[7]=w1.w;
        bv[8]=w2.x; bv[9]=w2.y; bv[10]=w2.z; bv[11]=w2.w;
        bv[12]=w3.x; bv[13]=w3.y; bv[14]=w3.z; bv[15]=w3.w;
        v16h B;
#pragma unroll
        for (int s = 0; s < 16; ++s) {
          float q = fmaf(bv[s], scale, beta);
          B[s] = (q == tgt) ? (_Float16)1.0f : (_Float16)0.0f;
        }
        acc[o] = __builtin_amdgcn_wmma_f32_16x16x32_f16(
            false, A, false, B, (short)0, acc[o], false, false);
      }
    } else {
      // ---- A fragment + hitmask (needed for pad counts) ----
      v16h A;
      uint32_t hitmask = 0;
#pragma unroll
      for (int s = 0; s < 16; ++s) {
        float q  = fmaf(av[s], scale, beta);
        bool hit = (q == tgt);
        A[s] = hit ? (_Float16)1.0f : (_Float16)0.0f;
        hitmask |= hit ? (1u << s) : 0u;
      }
#pragma unroll
      for (int o = 0; o < 4; ++o) {
        const int  yy     = y + DY[o];
        const bool rowOK  = (yy >= 0) && (yy < IMG_H);
        const int  offlin = DY[o] * IMG_W + DX[o];
        float bv[16];
#pragma unroll
        for (int r = 0; r < 4; ++r) {
          const int loc = 8 * g + (r & 1) * 4 + (r >> 1) * 16;
          int idx = rowbase + loc + offlin;
          if (idx < 0) idx = 0;
          if (idx > NPIX - 4) idx = NPIX - 4;   // clamped; invalid lanes masked below
          floatx4 w = *(const floatx4*)(im + idx);
          bv[4 * r + 0] = w.x; bv[4 * r + 1] = w.y;
          bv[4 * r + 2] = w.z; bv[4 * r + 3] = w.w;
        }
        v16h B;
#pragma unroll
        for (int s = 0; s < 16; ++s) {
          const int  xb  = x0 + 8 * g + (s & 7) + ((s >> 3) * 16);  // center x for this K slot
          const bool vOK = rowOK && ((unsigned)(xb + DX[o]) < (unsigned)IMG_W);
          float q = fmaf(bv[s], scale, beta);
          B[s] = (vOK && (q == tgt)) ? (_Float16)1.0f : (_Float16)0.0f;
        }
        acc[o] = __builtin_amdgcn_wmma_f32_16x16x32_f16(
            false, A, false, B, (short)0, acc[o], false, false);
        // pad-count Pc: centers (A layout: pixel 8g+(s&7)+16*(s>>3)) whose neighbor is invalid
#pragma unroll
        for (int s = 0; s < 16; ++s) {
          const int  xa  = x0 + 8 * g + (s & 7) + ((s >> 3) * 16);
          const bool vOK = rowOK && ((unsigned)(xa + DX[o]) < (unsigned)IMG_W);
          pc[o] += ((!vOK) && ((hitmask >> s) & 1u)) ? 1.0f : 0.0f;
        }
      }
    }
  }

  // ---- reduce wave partials to workspace ----
  // C/D layout: lane l, VGPR r  ->  element (M = r + 8*(l>=16), N = l&15)
#pragma unroll
  for (int o = 0; o < 4; ++o) {
    float* dst = nacc + ((img * NOFF + o) << 8);
#pragma unroll
    for (int r = 0; r < 8; ++r)
      atomicAdd(&dst[(r + 8 * g) * 16 + m], acc[o][r]);
    atomicAdd(&pcacc[(img * NOFF + o) * 16 + m], pc[o]);
  }
}

// ---------------- kernel 3: finalize (symmetrize + pad fix + contrast) ----------------
__global__ void finalize_kernel(const float* __restrict__ nacc,
                                const float* __restrict__ pcacc,
                                float* __restrict__ out) {
  const int t = threadIdx.x;                // t = img*4 + offset
  if (t >= NIMG * NOFF) return;
  constexpr int DY[4] = {0, -5, -5, -5};
  constexpr int DX[4] = {5,  5,  0, -5};
  const int o = t & 3;
  const float* N = nacc + (t << 8);
  const float* P = pcacc + (t << 4);
  const int ady = DY[o] < 0 ? -DY[o] : DY[o];
  const int adx = DX[o] < 0 ? -DX[o] : DX[o];
  const float Pt = (float)(IMG_H * IMG_W - (IMG_H - ady) * (IMG_W - adx));
  float S = 0.0f, ctr = 0.0f;
  for (int i = 0; i < 16; ++i) {
    for (int j = 0; j < 16; ++j) {
      float v = N[i * 16 + j] + N[j * 16 + i];       // M = N + N^T
      if (i + j == 18) {                             // PAD_VAL(20) - 2
        v += Pt - P[i] - P[j];
        if (i == j) v += Pt;
      }
      S += v;
      float d = (float)(i - j);
      ctr += d * d * v;
    }
  }
  out[t] = ctr / S;
}

// ---------------- launcher ----------------
extern "C" void kernel_launch(void* const* d_in, const int* in_sizes, int n_in,
                              void* d_out, int out_size, void* d_ws, size_t ws_size,
                              hipStream_t stream) {
  (void)in_sizes; (void)n_in; (void)out_size; (void)ws_size;
  const float* x = (const float*)d_in[0];
  float* out = (float*)d_out;

  uint8_t* ws = (uint8_t*)d_ws;
  uint32_t* keys = (uint32_t*)ws;                                  // 16 uints
  float* nacc  = (float*)(ws + 256);                               // 8*4*256 f32
  float* pcacc = (float*)(ws + 256 + NIMG * NOFF * 256 * sizeof(float)); // 8*4*16 f32

  init_ws_kernel <<<1, 256, 0, stream>>>(keys, nacc, pcacc);
  minmax_kernel  <<<NIMG * MMB, 256, 0, stream>>>(x, keys);
  glcm_wmma_kernel<<<NIMG * BPI, 256, 0, stream>>>(x, keys, nacc, pcacc);
  finalize_kernel<<<1, 32, 0, stream>>>(nacc, pcacc, out);
}
